// GELU2_5892695130299
// MI455X (gfx1250) — matrix operand order repaired and checked by reading, not compiled, not run on hardware
//
#include <hip/hip_runtime.h>
#include <math.h>

typedef __bf16 bf16_t;
typedef __attribute__((ext_vector_type(16))) __bf16 v16bf;
typedef __attribute__((ext_vector_type(8)))  float  v8f;

#define NTOK 16384   // B*T
#define DDIM 4096
#define KPROT 256

// workspace byte offsets (all >= 1KiB aligned)
#define OFF_PNT    0u         // bf16 [DDIM][KPROT]  = 2 MiB (transposed normalized protos)
#define OFF_PINV   2097152u   // f32 [KPROT]
#define OFF_XINV   2098176u   // f32 [NTOK]
#define OFF_MAXSIM 2163712u   // f32 [NTOK]
#define OFF_PART   2229248u   // f32 [64][DDIM] column-sum partials = 1 MiB
#define OFF_XMEAN  3277824u   // f32 [DDIM]
#define OFF_SCORE  3294208u   // f32 [KPROT]
#define OFF_KSTAR  3295232u   // int

__device__ __forceinline__ float sigmoid_(float v) { return 1.0f / (1.0f + expf(-v)); }

__device__ __forceinline__ float gelu_tanh_(float y) {
    const float c = 0.7978845608028654f;
    float y3 = y * y * y;
    return 0.5f * y * (1.0f + tanhf(c * (y + 0.044715f * y3)));
}

// ---- K1: normalize prototypes, write inv-norms + transposed bf16 pnT[d][k] ----
__global__ void k_proto_norm(const float* __restrict__ emas, float* __restrict__ pinv,
                             bf16_t* __restrict__ pnt) {
    __shared__ float red[256];
    int k = blockIdx.x, tid = threadIdx.x;
    const float* row = emas + (size_t)k * DDIM;
    float s = 0.f;
    for (int d = tid; d < DDIM; d += 256) { float v = row[d]; s += v * v; }
    red[tid] = s; __syncthreads();
    for (int o = 128; o > 0; o >>= 1) { if (tid < o) red[tid] += red[tid + o]; __syncthreads(); }
    float inv = 1.0f / fmaxf(sqrtf(red[0]), 1e-12f);
    if (tid == 0) pinv[k] = inv;
    for (int d = tid; d < DDIM; d += 256)
        pnt[(size_t)d * KPROT + k] = (bf16_t)(row[d] * inv);
}

// ---- K2: per-token inverse L2 norm ----
__global__ void k_token_norm(const float* __restrict__ x, float* __restrict__ xinv) {
    __shared__ float red[256];
    int t = blockIdx.x, tid = threadIdx.x;
    const float4* row = (const float4*)(x + (size_t)t * DDIM);
    float s = 0.f;
    for (int i = tid; i < DDIM / 4; i += 256) {
        float4 v = row[i];
        s += v.x * v.x + v.y * v.y + v.z * v.z + v.w * v.w;
    }
    red[tid] = s; __syncthreads();
    for (int o = 128; o > 0; o >>= 1) { if (tid < o) red[tid] += red[tid + o]; __syncthreads(); }
    if (tid == 0) xinv[t] = 1.0f / fmaxf(sqrtf(red[0]), 1e-12f);
}

// ---- K3/K4: deterministic two-stage column sum for x_mean ----
__global__ void k_colsum_part(const float* __restrict__ x, float* __restrict__ part) {
    int d = blockIdx.x * 256 + threadIdx.x;   // gridDim.x = 16
    int tc = blockIdx.y;                      // gridDim.y = 64 (256 tokens each)
    const float* p = x + (size_t)tc * 256 * DDIM + d;
    float s = 0.f;
    for (int t = 0; t < 256; ++t) s += p[(size_t)t * DDIM];
    part[(size_t)tc * DDIM + d] = s;
}

__global__ void k_colsum_fin(const float* __restrict__ part, float* __restrict__ xmean) {
    int d = blockIdx.x * 256 + threadIdx.x;
    float s = 0.f;
    for (int c = 0; c < 64; ++c) s += part[(size_t)c * DDIM + d];
    xmean[d] = s * (1.0f / (float)NTOK);
}

// ---- K5: WMMA GEMM sim = x_norm @ p_normT, fused max over K -> maxsim[t] ----
// One wave per 32 tokens. A: bf16 from f32 x * inv-norm (in-register convert).
// B: strided v16bf gather from pnT (lane = proto, elements = 16 d values, same k order as A).
__global__ void __launch_bounds__(32) k_sim_max(const float* __restrict__ x,
                                                const float* __restrict__ xinv,
                                                const bf16_t* __restrict__ pnt,
                                                float* __restrict__ maxsim) {
    int lane = threadIdx.x;
    int half = lane >> 4, lm = lane & 15;
    int t0 = blockIdx.x * 32;

    const float* xrow0 = x + (size_t)(t0 + lm) * DDIM;
    const float* xrow1 = x + (size_t)(t0 + 16 + lm) * DDIM;
    float inv0 = xinv[t0 + lm];
    float inv1 = xinv[t0 + 16 + lm];

    v8f acc[2][16] = {};

    for (int dc = 0; dc < DDIM; dc += 32) {
        int d0 = dc + half * 8;
        // prefetch next chunk of the streaming rows (global_prefetch_b8)
        __builtin_prefetch(xrow0 + d0 + 64, 0, 3);
        __builtin_prefetch(xrow1 + d0 + 64, 0, 3);

        const float4* pa0 = (const float4*)(xrow0 + d0);
        const float4* pa1 = (const float4*)(xrow1 + d0);
        float av0[16], av1[16];
        float4 q;
        q = pa0[0]; av0[0] = q.x; av0[1] = q.y; av0[2]  = q.z; av0[3]  = q.w;
        q = pa0[1]; av0[4] = q.x; av0[5] = q.y; av0[6]  = q.z; av0[7]  = q.w;
        q = pa0[4]; av0[8] = q.x; av0[9] = q.y; av0[10] = q.z; av0[11] = q.w;
        q = pa0[5]; av0[12]= q.x; av0[13]= q.y; av0[14] = q.z; av0[15] = q.w;
        q = pa1[0]; av1[0] = q.x; av1[1] = q.y; av1[2]  = q.z; av1[3]  = q.w;
        q = pa1[1]; av1[4] = q.x; av1[5] = q.y; av1[6]  = q.z; av1[7]  = q.w;
        q = pa1[4]; av1[8] = q.x; av1[9] = q.y; av1[10] = q.z; av1[11] = q.w;
        q = pa1[5]; av1[12]= q.x; av1[13]= q.y; av1[14] = q.z; av1[15] = q.w;

        v16bf af0, af1;
#pragma unroll
        for (int e = 0; e < 16; ++e) {
            af0[e] = (bf16_t)(av0[e] * inv0);
            af1[e] = (bf16_t)(av1[e] * inv1);
        }

        const bf16_t* bbase = pnt + (size_t)(dc + half * 8) * KPROT + lm;
#pragma unroll
        for (int pc = 0; pc < 16; ++pc) {
            v16bf bf;
#pragma unroll
            for (int e = 0; e < 16; ++e)
                bf[e] = bbase[(size_t)(e < 8 ? e : e + 8) * KPROT + pc * 16];
            acc[0][pc] = __builtin_amdgcn_wmma_f32_16x16x32_bf16(
                false, af0, false, bf, (short)0, acc[0][pc], false, false);
            acc[1][pc] = __builtin_amdgcn_wmma_f32_16x16x32_bf16(
                false, af1, false, bf, (short)0, acc[1][pc], false, false);
        }
    }

    // C/D layout: lane&15 = proto col, VGPR v -> token m = v + 8*(lane>>4)
#pragma unroll
    for (int mt = 0; mt < 2; ++mt) {
#pragma unroll
        for (int v = 0; v < 8; ++v) {
            float m = acc[mt][0][v];
#pragma unroll
            for (int pc = 1; pc < 16; ++pc) m = fmaxf(m, acc[mt][pc][v]);
            m = fmaxf(m, __shfl_xor(m, 1));
            m = fmaxf(m, __shfl_xor(m, 2));
            m = fmaxf(m, __shfl_xor(m, 4));
            m = fmaxf(m, __shfl_xor(m, 8));
            if (lm == 0) maxsim[t0 + mt * 16 + half * 8 + v] = m;
        }
    }
}

// ---- K6: score[k] = dot(p_norm[k], x_mean) (argmax-equivalent scoring) ----
__global__ void k_score(const float* __restrict__ emas, const float* __restrict__ xmean,
                        const float* __restrict__ pinv, float* __restrict__ score) {
    __shared__ float red[256];
    int k = blockIdx.x, tid = threadIdx.x;
    const float* row = emas + (size_t)k * DDIM;
    float s = 0.f;
    for (int d = tid; d < DDIM; d += 256) s += row[d] * xmean[d];
    red[tid] = s; __syncthreads();
    for (int o = 128; o > 0; o >>= 1) { if (tid < o) red[tid] += red[tid + o]; __syncthreads(); }
    if (tid == 0) score[k] = red[0] * pinv[k];
}

// ---- K7: argmax (first-max index, deterministic) ----
__global__ void k_argmax(const float* __restrict__ score, int* __restrict__ kstar) {
    float best = score[0]; int bi = 0;
    for (int k = 1; k < KPROT; ++k) { float v = score[k]; if (v > best) { best = v; bi = k; } }
    *kstar = bi;
}

// ---- K8: new_emas (copy with one EMA-updated row) ----
__global__ void k_new_emas(const float* __restrict__ emas, const float* __restrict__ xmean,
                           const int* __restrict__ kstar, const float* __restrict__ logit_decay,
                           float* __restrict__ out) {
    int idx = blockIdx.x * 256 + threadIdx.x;
    int k = idx >> 12, d = idx & (DDIM - 1);
    float dd = sigmoid_(logit_decay[0]);
    float v = emas[idx];
    if (k == *kstar) v = dd * v + (1.0f - dd) * xmean[d];
    out[idx] = v;
}

// ---- K9: out = gelu(x * scale[t]) , float4-vectorized ----
__global__ void k_out(const float* __restrict__ x, const float* __restrict__ maxsim,
                      const float* __restrict__ lt, const float* __restrict__ lb,
                      const float* __restrict__ ld, float* __restrict__ out) {
    size_t i4 = (size_t)blockIdx.x * 256 + threadIdx.x;   // one float4 per thread
    float tau = expf(lt[0]);
    float ad = sigmoid_(lb[0]) * sigmoid_(ld[0]);
    size_t t = i4 >> 10;                                  // 1024 float4 per token row
    float scale = 1.0f - ad + ad * expf(-tau * maxsim[t]);
    float4 v = ((const float4*)x)[i4];
    float4 r;
    r.x = gelu_tanh_(v.x * scale);
    r.y = gelu_tanh_(v.y * scale);
    r.z = gelu_tanh_(v.z * scale);
    r.w = gelu_tanh_(v.w * scale);
    ((float4*)out)[i4] = r;
}

extern "C" void kernel_launch(void* const* d_in, const int* in_sizes, int n_in,
                              void* d_out, int out_size, void* d_ws, size_t ws_size,
                              hipStream_t stream) {
    const float* x           = (const float*)d_in[0];
    const float* emas        = (const float*)d_in[1];
    const float* log_tau     = (const float*)d_in[2];
    const float* log_blend   = (const float*)d_in[3];
    const float* logit_decay = (const float*)d_in[4];

    float* out      = (float*)d_out;
    float* out_emas = out + (size_t)NTOK * DDIM;

    char* ws = (char*)d_ws;
    bf16_t* pnt   = (bf16_t*)(ws + OFF_PNT);
    float* pinv   = (float*)(ws + OFF_PINV);
    float* xinv   = (float*)(ws + OFF_XINV);
    float* maxsim = (float*)(ws + OFF_MAXSIM);
    float* part   = (float*)(ws + OFF_PART);
    float* xmean  = (float*)(ws + OFF_XMEAN);
    float* score  = (float*)(ws + OFF_SCORE);
    int*   kstar  = (int*)(ws + OFF_KSTAR);

    k_proto_norm<<<KPROT, 256, 0, stream>>>(emas, pinv, pnt);
    k_token_norm<<<NTOK, 256, 0, stream>>>(x, xinv);
    k_colsum_part<<<dim3(16, 64), 256, 0, stream>>>(x, part);
    k_colsum_fin<<<16, 256, 0, stream>>>(part, xmean);
    k_sim_max<<<NTOK / 32, 32, 0, stream>>>(x, xinv, pnt, maxsim);
    k_score<<<KPROT, 256, 0, stream>>>(emas, xmean, pinv, score);
    k_argmax<<<1, 1, 0, stream>>>(score, kstar);
    k_new_emas<<<(KPROT * DDIM) / 256, 256, 0, stream>>>(emas, xmean, kstar, logit_decay, out_emas);
    k_out<<<(size_t)NTOK * DDIM / 4 / 256, 256, 0, stream>>>(x, maxsim, log_tau, log_blend,
                                                             logit_decay, out);
}
